// BertAttention_56650618634988
// MI455X (gfx1250) — matrix operand order repaired and checked by reading, not compiled, not run on hardware
//
#include <hip/hip_runtime.h>

// ---------------- problem constants ----------------
#define SB   2048            // sequence length
#define HD_  1024            // hidden
#define NH_  16              // heads
#define DH   64              // head dim
#define BB   4               // batch
#define ROWS (BB * SB)       // 8192 token rows

typedef _Float16 v4h  __attribute__((ext_vector_type(4)));
typedef _Float16 v8h  __attribute__((ext_vector_type(8)));
typedef _Float16 v16h __attribute__((ext_vector_type(16)));
typedef float    v8f  __attribute__((ext_vector_type(8)));
typedef unsigned int v4u __attribute__((ext_vector_type(4)));
typedef int      v4i  __attribute__((ext_vector_type(4)));
typedef int      v8i  __attribute__((ext_vector_type(8)));

static __device__ inline v16h cat8(v8h a, v8h b) {
  return __builtin_shufflevector(a, b, 0,1,2,3,4,5,6,7,8,9,10,11,12,13,14,15);
}

// A fragment: 16 rows x 32 K of a row-major matrix (leading dim ld), K slice at k.
// Lane L: row L&15; halves 0..15 = contiguous K run hi*16 + 0..15.
static __device__ inline v16h load_afrag(const _Float16* __restrict__ base,
                                         int row0, int ld, int k, int lane) {
  const _Float16* p = base + (size_t)(row0 + (lane & 15)) * ld + k + ((lane >> 4) << 4);
  return cat8(*(const v8h*)p, *(const v8h*)(p + 8));
}

// B fragment for B = M^T (B[k][n] = M[n][k]), M row-major, leading dim ld.
// Lane L: col n0+(L&15); halves = contiguous K run hi*16 + 0..15.
static __device__ inline v16h load_bfragT(const _Float16* __restrict__ base,
                                          int n0, int ld, int k, int lane) {
  const _Float16* p = base + (size_t)(n0 + (lane & 15)) * ld + k + ((lane >> 4) << 4);
  return cat8(*(const v8h*)p, *(const v8h*)(p + 8));
}

#define WMMA_F16(a, b, c) \
  __builtin_amdgcn_wmma_f32_16x16x32_f16(false, (a), false, (b), (short)0, (c), false, false)

// ---------------- TDM: 2D f16 tile (tile_d0 x tile_d1, row stride 'stride' halves)
// from global 'src' into LDS at 'ldsoff'. Per CDNA5 ISA D# layout.
// This toolchain exposes the 6-arg builtin: (v4u, v8i, v4i, v4i, v8i, i32 cpol).
static __device__ inline void tdm_load_tile_f16(const _Float16* src, unsigned int ldsoff,
                                                int tile_d0, int tile_d1, int stride) {
#if __has_builtin(__builtin_amdgcn_tensor_load_to_lds)
  const unsigned long long ga = (unsigned long long)(size_t)src;
  v4u g0;
  g0[0] = 1u;                                   // count=1, is_restore=0, gather=0
  g0[1] = ldsoff;                               // lds_addr (bytes)
  g0[2] = (unsigned int)ga;                     // global_addr[31:0]
  g0[3] = ((unsigned int)(ga >> 32) & 0x01FFFFFFu) | (2u << 30);  // addr[56:32] | type=2
  v8i g1;
  g1[0] = (int)(1u << 16);                      // wg_mask=0, data_size=1 (2 bytes)
  g1[1] = 0;                                    // atomic_barrier=0, tensor_dim0 lo16 (=0)
  g1[2] = (int)0x10u;                           // tensor_dim0 = 1<<20 (hi16), dim1 lo16=0
  g1[3] = (int)(0x10u | ((unsigned)tile_d0 << 16)); // tensor_dim1 = 1<<20, tile_dim0
  g1[4] = (int)(unsigned)tile_d1;               // tile_dim1, tile_dim2=0
  g1[5] = stride;                               // tensor_dim0_stride lo32 (halves)
  g1[6] = 0;                                    // stride0 hi, stride1 lo
  g1[7] = 0;
  v4i g2 = {0, 0, 0, 0};                        // group2: unused (2D tensor)
  v4i g3 = {0, 0, 0, 0};                        // group3: unused
  v8i g4 = {0, 0, 0, 0, 0, 0, 0, 0};            // extra group (6-arg form), unused
  __builtin_amdgcn_tensor_load_to_lds(g0, g1, g2, g3, g4, 0);
#else
  (void)src; (void)ldsoff; (void)tile_d0; (void)tile_d1; (void)stride;
#endif
}

// ---------------- fp32 -> f16 convert ----------------
__global__ __launch_bounds__(256) void cvt_f32_f16(const float* __restrict__ in,
                                                   _Float16* __restrict__ out, int n) {
  int i = (blockIdx.x * 256 + threadIdx.x) * 4;
  if (i + 3 < n) {
    float4 v = *(const float4*)(in + i);
    v4h o;
    o[0] = (_Float16)v.x; o[1] = (_Float16)v.y;
    o[2] = (_Float16)v.z; o[3] = (_Float16)v.w;
    *(v4h*)(out + i) = o;
  }
}

// ---------------- GEMM: out = X @ W^T + bias (+residual) ----------------
// X: [ROWS,1024] f16; W: [1024,1024] f16 row-major.
// B (=W^T) tiles staged through LDS by the Tensor Data Mover, double-buffered.
// mode 0: out16 f16 [B,NH,S,DH] ; mode 2: out16 f16 [B,NH,DH,S] ; mode 3: f32 + residual.
// Block 256 thr = 8 waves; wave: 32 rows x 64 cols (8 WMMA / k-step, B reused 2x);
// block: 256 rows x 64 cols; grid (16, 32).
__global__ __launch_bounds__(256) void gemm_xwt_kernel(
    const _Float16* __restrict__ X, const _Float16* __restrict__ W,
    const float* __restrict__ bias, const float* __restrict__ residual,
    _Float16* __restrict__ out16, float* __restrict__ out32, int mode) {
  __shared__ _Float16 btile[2][64 * 32];   // 2 x 4KB: [col-in-tile][k-in-step]

  const int lane = threadIdx.x & 31;
  const int wave = threadIdx.x >> 5;
  const int r0 = blockIdx.y * 256 + wave * 32;
  const int c0 = blockIdx.x * 64;
  const int nloc = lane & 15;
  const int hi   = lane >> 4;
  const int NIT = HD_ / 32;

  if (wave == 0) {
    // rows c0..c0+63 of W, k-slice [0,32): tile 32 halves x 64 rows, stride 1024
    tdm_load_tile_f16(W + (size_t)c0 * HD_, (unsigned int)(size_t)&btile[0][0],
                      32, 64, HD_);
  }

  v8f acc[2][4] = {};
  v16h a_cur0 = load_afrag(X, r0,      HD_, 0, lane);
  v16h a_cur1 = load_afrag(X, r0 + 16, HD_, 0, lane);

  for (int kk = 0; kk < NIT; ++kk) {
    const int k = kk * 32;
    if (wave == 0) __builtin_amdgcn_s_wait_tensorcnt(0);
    __syncthreads();                         // btile[kk&1] ready; prev reads fenced
    if (wave == 0 && kk + 1 < NIT) {
      tdm_load_tile_f16(W + (size_t)c0 * HD_ + (k + 32),
                        (unsigned int)(size_t)&btile[(kk + 1) & 1][0], 32, 64, HD_);
    }
    // prefetch next A fragments (benign over-read on last step: next ws buffer)
    v16h a_nxt0 = load_afrag(X, r0,      HD_, k + 32, lane);
    v16h a_nxt1 = load_afrag(X, r0 + 16, HD_, k + 32, lane);

    // load ALL B fragments first so ds latency is paid once, then batch the WMMAs
    const _Float16* bt = &btile[kk & 1][0];
    v16h bfr[4];
#pragma unroll
    for (int nt = 0; nt < 4; ++nt) {
      const _Float16* p = bt + (nt * 16 + nloc) * 32 + (hi << 4);
      bfr[nt] = cat8(*(const v8h*)p, *(const v8h*)(p + 8));
    }
#pragma unroll
    for (int nt = 0; nt < 4; ++nt) {
      acc[0][nt] = WMMA_F16(a_cur0, bfr[nt], acc[0][nt]);
      acc[1][nt] = WMMA_F16(a_cur1, bfr[nt], acc[1][nt]);
    }
    a_cur0 = a_nxt0;
    a_cur1 = a_nxt1;
  }

#pragma unroll
  for (int m = 0; m < 2; ++m) {
#pragma unroll
    for (int nt = 0; nt < 4; ++nt) {
      const int c  = c0 + nt * 16 + nloc;
      const float bv = bias[c];
#pragma unroll
      for (int i = 0; i < 8; ++i) {
        const int r = r0 + m * 16 + i + 8 * hi;
        float v = acc[m][nt][i] + bv;
        if (mode == 3) {
          out32[(size_t)r * HD_ + c] = v + residual[(size_t)r * HD_ + c];
        } else {
          const int b = r >> 11, s = r & (SB - 1);
          const int hh = c >> 6, d = c & (DH - 1);
          if (mode == 0)
            out16[(((size_t)b * NH_ + hh) * SB + s) * DH + d] = (_Float16)v;
          else
            out16[(((size_t)b * NH_ + hh) * DH + d) * SB + s] = (_Float16)v;
        }
      }
    }
  }
}

// ---------------- flash attention ----------------
// Qh,Kh: [B,NH,S,DH] f16; Vt: [B,NH,DH,S] f16; mask: [B,1,1,S] f32.
// ctx out: [B,S,H] f16. One wave = 16 query rows of one head, K/V streamed 32 keys/step,
// K-fragments software-pipelined one step ahead; V loads overlap softmax + LDS transpose.
__global__ __launch_bounds__(256) void flash_attn_kernel(
    const _Float16* __restrict__ Qh, const _Float16* __restrict__ Kh,
    const _Float16* __restrict__ Vt, const float* __restrict__ mask,
    _Float16* __restrict__ ctx) {
  __shared__ _Float16 plds[8][16 * 32];

  const int lane = threadIdx.x & 31;
  const int wave = threadIdx.x >> 5;
  const int tile = blockIdx.x * 8 + wave;     // 8192 wave-tiles
  const int q0 = (tile & (SB / 16 - 1)) * 16;
  const int bh = tile >> 7;                   // b*NH + h
  const int b  = bh >> 4;
  const int hh = bh & (NH_ - 1);

  const _Float16* Qb = Qh + (size_t)bh * SB * DH;
  const _Float16* Kb = Kh + (size_t)bh * SB * DH;
  const _Float16* Vb = Vt + (size_t)bh * DH * SB;
  const float*    mb = mask + (size_t)b * SB;

  const v16h qa0 = load_afrag(Qb, q0, DH, 0, lane);
  const v16h qa1 = load_afrag(Qb, q0, DH, 32, lane);

  const int nloc = lane & 15;
  const int hi   = lane >> 4;

  float m_i[8], l_i[8];
#pragma unroll
  for (int i = 0; i < 8; ++i) { m_i[i] = -1e30f; l_i[i] = 0.0f; }
  v8f o[4] = {};

  _Float16* myp = &plds[wave][0];

  // prologue: K fragments for j = 0
  v16h kb[4];
  kb[0] = load_bfragT(Kb, 0, DH, 0, lane);
  kb[1] = load_bfragT(Kb, 0, DH, 32, lane);
  kb[2] = load_bfragT(Kb, 16, DH, 0, lane);
  kb[3] = load_bfragT(Kb, 16, DH, 32, lane);

  for (int j = 0; j < SB; j += 32) {
    __builtin_prefetch(Kb + (size_t)(j + 64) * DH, 0, 1);

    v8f s0 = {}, s1 = {};
    s0 = WMMA_F16(qa0, kb[0], s0);
    s0 = WMMA_F16(qa1, kb[1], s0);
    s1 = WMMA_F16(qa0, kb[2], s1);
    s1 = WMMA_F16(qa1, kb[3], s1);

    // pipeline: K fragments for j+32 (tail over-read lands in allocated Vt buffer)
    kb[0] = load_bfragT(Kb, j + 32, DH, 0, lane);
    kb[1] = load_bfragT(Kb, j + 32, DH, 32, lane);
    kb[2] = load_bfragT(Kb, j + 48, DH, 0, lane);
    kb[3] = load_bfragT(Kb, j + 48, DH, 32, lane);
    // V fragments for this step: latency overlaps softmax + LDS transpose below
    v16h vb0 = load_bfragT(Vb, 0,  SB, j, lane);
    v16h vb1 = load_bfragT(Vb, 16, SB, j, lane);
    v16h vb2 = load_bfragT(Vb, 32, SB, j, lane);
    v16h vb3 = load_bfragT(Vb, 48, SB, j, lane);

    const float mk0 = mb[j + nloc];
    const float mk1 = mb[j + 16 + nloc];

    float alpha[8];
#pragma unroll
    for (int i = 0; i < 8; ++i) {
      s0[i] = s0[i] * 0.125f + mk0;   // 1/sqrt(64)
      s1[i] = s1[i] * 0.125f + mk1;
      float mx = fmaxf(s0[i], s1[i]);
#pragma unroll
      for (int off = 8; off >= 1; off >>= 1) mx = fmaxf(mx, __shfl_xor(mx, off));
      const float mnew = fmaxf(m_i[i], mx);
      alpha[i] = __expf(m_i[i] - mnew);
      m_i[i] = mnew;
      s0[i] = __expf(s0[i] - mnew);
      s1[i] = __expf(s1[i] - mnew);
      float rs = s0[i] + s1[i];
#pragma unroll
      for (int off = 8; off >= 1; off >>= 1) rs += __shfl_xor(rs, off);
      l_i[i] = l_i[i] * alpha[i] + rs;
    }

    // P (C layout) -> LDS -> A-fragment layout (same-wave, DS in-order + explicit wait)
#pragma unroll
    for (int i = 0; i < 8; ++i) {
      const int row = i + 8 * hi;
      myp[row * 32 + nloc]      = (_Float16)s0[i];
      myp[row * 32 + 16 + nloc] = (_Float16)s1[i];
    }
    asm volatile("s_wait_dscnt 0" ::: "memory");
    v16h pa;
    {
      const _Float16* p = myp + (size_t)nloc * 32 + (hi << 4);
      pa = cat8(*(const v8h*)p, *(const v8h*)(p + 8));
    }

#pragma unroll
    for (int nt = 0; nt < 4; ++nt) {
#pragma unroll
      for (int i = 0; i < 8; ++i) o[nt][i] *= alpha[i];
    }
    o[0] = WMMA_F16(pa, vb0, o[0]);
    o[1] = WMMA_F16(pa, vb1, o[1]);
    o[2] = WMMA_F16(pa, vb2, o[2]);
    o[3] = WMMA_F16(pa, vb3, o[3]);
  }

  float inv[8];
#pragma unroll
  for (int i = 0; i < 8; ++i) inv[i] = 1.0f / l_i[i];
  _Float16* cb = ctx + ((size_t)b * SB + q0) * HD_ + hh * DH;
#pragma unroll
  for (int nt = 0; nt < 4; ++nt) {
#pragma unroll
    for (int i = 0; i < 8; ++i) {
      const int row = i + 8 * hi;
      cb[(size_t)row * HD_ + nt * 16 + nloc] = (_Float16)(o[nt][i] * inv[i]);
    }
  }
}

// ---------------- LayerNorm (TF eps inside sqrt) ----------------
__global__ __launch_bounds__(256) void layernorm_kernel(
    const float* __restrict__ h, const float* __restrict__ w,
    const float* __restrict__ bb, float* __restrict__ out) {
  __shared__ float red[2][8];
  const int row = blockIdx.x;
  const float* hr = h + (size_t)row * HD_;
  const int t = threadIdx.x;
  float4 v = *(const float4*)(hr + t * 4);
  float sum = v.x + v.y + v.z + v.w;
  float sq  = v.x * v.x + v.y * v.y + v.z * v.z + v.w * v.w;
#pragma unroll
  for (int off = 16; off >= 1; off >>= 1) {
    sum += __shfl_xor(sum, off);
    sq  += __shfl_xor(sq, off);
  }
  const int wv = t >> 5, ln = t & 31;
  if (ln == 0) { red[0][wv] = sum; red[1][wv] = sq; }
  __syncthreads();
  float ts = 0.f, tq = 0.f;
#pragma unroll
  for (int i = 0; i < 8; ++i) { ts += red[0][i]; tq += red[1][i]; }
  const float mean = ts * (1.0f / HD_);
  const float var  = tq * (1.0f / HD_) - mean * mean;
  const float rstd = rsqrtf(var + 1e-12f);
  float4 wvv = *(const float4*)(w + t * 4);
  float4 bvv = *(const float4*)(bb + t * 4);
  float4 ov;
  ov.x = wvv.x * (v.x - mean) * rstd + bvv.x;
  ov.y = wvv.y * (v.y - mean) * rstd + bvv.y;
  ov.z = wvv.z * (v.z - mean) * rstd + bvv.z;
  ov.w = wvv.w * (v.w - mean) * rstd + bvv.w;
  *(float4*)(out + (size_t)row * HD_ + t * 4) = ov;
}

// ---------------- launch ----------------
extern "C" void kernel_launch(void* const* d_in, const int* in_sizes, int n_in,
                              void* d_out, int out_size, void* d_ws, size_t ws_size,
                              hipStream_t stream) {
  (void)in_sizes; (void)n_in; (void)out_size; (void)ws_size;
  const float* x    = (const float*)d_in[0];
  const float* am   = (const float*)d_in[1];
  const float* Wq   = (const float*)d_in[2];
  const float* bq   = (const float*)d_in[3];
  const float* Wk   = (const float*)d_in[4];
  const float* bk   = (const float*)d_in[5];
  const float* Wv   = (const float*)d_in[6];
  const float* bv   = (const float*)d_in[7];
  const float* Wo   = (const float*)d_in[8];
  const float* bo   = (const float*)d_in[9];
  const float* lnw  = (const float*)d_in[10];
  const float* lnb  = (const float*)d_in[11];
  float* out = (float*)d_out;

  const size_t NW = (size_t)HD_ * HD_;   // 1M weight elems
  const size_t NX = (size_t)ROWS * HD_;  // 8M activation elems
  _Float16* Whq  = (_Float16*)d_ws;
  _Float16* Whk  = Whq + NW;
  _Float16* Whv  = Whk + NW;
  _Float16* Who  = Whv + NW;
  _Float16* xh   = Who + NW;
  _Float16* Qh   = xh + NX;
  _Float16* Kh   = Qh + NX;
  _Float16* Vt   = Kh + NX;
  _Float16* ctxh = Vt + NX;
  float*    hbuf = (float*)(ctxh + NX);

  const int gW = (int)(NW / 4 / 256);
  const int gX = (int)(NX / 4 / 256);
  cvt_f32_f16<<<gW, 256, 0, stream>>>(Wq, Whq, (int)NW);
  cvt_f32_f16<<<gW, 256, 0, stream>>>(Wk, Whk, (int)NW);
  cvt_f32_f16<<<gW, 256, 0, stream>>>(Wv, Whv, (int)NW);
  cvt_f32_f16<<<gW, 256, 0, stream>>>(Wo, Who, (int)NW);
  cvt_f32_f16<<<gX, 256, 0, stream>>>(x, xh, (int)NX);

  dim3 gg(HD_ / 64, ROWS / 256);
  gemm_xwt_kernel<<<gg, 256, 0, stream>>>(xh, Whq, bq, nullptr, Qh, nullptr, 0);
  gemm_xwt_kernel<<<gg, 256, 0, stream>>>(xh, Whk, bk, nullptr, Kh, nullptr, 0);
  gemm_xwt_kernel<<<gg, 256, 0, stream>>>(xh, Whv, bv, nullptr, Vt, nullptr, 2);

  flash_attn_kernel<<<(BB * NH_ * (SB / 16)) / 8, 256, 0, stream>>>(Qh, Kh, Vt, am, ctxh);

  gemm_xwt_kernel<<<gg, 256, 0, stream>>>(ctxh, Who, bo, x, nullptr, hbuf, 3);

  layernorm_kernel<<<ROWS, 256, 0, stream>>>(hbuf, lnw, lnb, out);
}
